// WindowMultiHeadAttention_65180423685378
// MI455X (gfx1250) — matrix operand order repaired and checked by reading, not compiled, not run on hardware
//
#include <hip/hip_runtime.h>
#include <math.h>

typedef __attribute__((ext_vector_type(16))) _Float16 v16h;
typedef __attribute__((ext_vector_type(8)))  _Float16 v8h;
typedef __attribute__((ext_vector_type(8)))  float    v8f;

#define N_TOK   64
#define C_      192
#define C3_     576
#define H_      6
#define HD_     32
#define NMASK_  64
#define BW_     4096

#define LDA_    200   // halfs: tokens, later attention output (row bytes 400, 16B-aligned)
#define LDQ_    584   // halfs: q,k (row bytes 1168, 16B-aligned)
#define LDVT_   72    // halfs: v transposed [hd][token] (row bytes 144, 16B-aligned)
#define LDF_    196   // floats: final result (aliases smQ)

// ws layout (bytes)
#define WS_BIAS_OFF   0
#define WS_BIAS_SZ    (H_ * N_TOK * N_TOK * 4)                 // 98304
#define WS_WQ_OFF     (WS_BIAS_OFF + WS_BIAS_SZ)
#define WS_WQ_SZ      (C_ * C3_ * 2)                           // 221184
#define WS_WP_OFF     (WS_WQ_OFF + WS_WQ_SZ)
#define WS_WP_SZ      (C_ * C_ * 2)                            // 73728
#define WS_COMBO_OFF  (WS_WP_OFF + WS_WP_SZ)                   // 393216
#define WS_COMBO_SZ   (NMASK_ * H_ * N_TOK * N_TOK * 4)        // 6291456
#define WS_TOTAL      (WS_COMBO_OFF + WS_COMBO_SZ)             // 6684672

// ---- fragment loaders (16x16x32 f16 WMMA layouts, wave32) -----------------
// A 16x32: lane l holds row m0+(l&15); halves [16*(l>>4)..+15] (contiguous per-lane run)
__device__ __forceinline__ v16h ld_a16(const _Float16* M, int ld, int m0, int k0, int lane) {
  const _Float16* p = M + (m0 + (lane & 15)) * ld + k0 + ((lane >> 4) << 4);
  v8h lo = *(const v8h*)p;
  v8h hi = *(const v8h*)(p + 8);
  v16h r;
  #pragma unroll
  for (int j = 0; j < 8; ++j) { r[j] = lo[j]; r[8 + j] = hi[j]; }
  return r;
}
// B 32x16 transposed source: B[k][n] = M[(n0+n)*ld + k0+k]  (contiguous per-lane run)
__device__ __forceinline__ v16h ld_bt16(const _Float16* M, int ld, int k0, int n0, int lane) {
  const _Float16* p = M + (n0 + (lane & 15)) * ld + k0 + ((lane >> 4) << 4);
  v8h lo = *(const v8h*)p;
  v8h hi = *(const v8h*)(p + 8);
  v16h r;
  #pragma unroll
  for (int j = 0; j < 8; ++j) { r[j] = lo[j]; r[8 + j] = hi[j]; }
  return r;
}
// B 32x16 from row-major f32 global (fallback when no pack workspace)
__device__ __forceinline__ v16h ld_b32g(const float* M, int ld, int k0, int n0, int lane) {
  v16h r;
  const float* p = M + (size_t)(k0 + ((lane >> 4) << 4)) * ld + n0 + (lane & 15);
  #pragma unroll
  for (int j = 0; j < 16; ++j) r[j] = (_Float16)p[(size_t)j * ld];
  return r;
}

// ---- kernel 1: meta-network relative position bias (constant over windows) ---
__global__ void crpb_bias_kernel(const float* __restrict__ mW1, const float* __restrict__ mb1,
                                 const float* __restrict__ mW2, const float* __restrict__ mb2,
                                 float* __restrict__ biasg) {
  int p = blockIdx.x * blockDim.x + threadIdx.x;   // m*64+n
  if (p >= N_TOK * N_TOK) return;
  int m = p >> 6, n = p & 63;
  float d0 = (float)(m >> 3) - (float)(n >> 3);
  float d1 = (float)(m & 7) - (float)(n & 7);
  float r0 = (d0 > 0.f ? 1.f : (d0 < 0.f ? -1.f : 0.f)) * log1pf(fabsf(d0));
  float r1 = (d1 > 0.f ? 1.f : (d1 < 0.f ? -1.f : 0.f)) * log1pf(fabsf(d1));
  float o[H_];
  #pragma unroll
  for (int h = 0; h < H_; ++h) o[h] = mb2[h];
  for (int j = 0; j < 256; ++j) {
    float hj = fmaxf(r0 * mW1[j] + r1 * mW1[256 + j] + mb1[j], 0.f);
    #pragma unroll
    for (int h = 0; h < H_; ++h) o[h] += hj * mW2[j * H_ + h];
  }
  #pragma unroll
  for (int h = 0; h < H_; ++h) biasg[h * (N_TOK * N_TOK) + p] = o[h];
}

// ---- kernel 2: repack W_qkv / W_proj into f16 B-fragment tiles -------------
// tile tt, lane l, half j  ->  B[k][n], k = kt*32 + 16*(l>>4) + j, n = nt*16 + (l&15)
__global__ void pack_weights_kernel(const float* __restrict__ Wqkv,
                                    const float* __restrict__ Wproj,
                                    _Float16* __restrict__ wq, _Float16* __restrict__ wp) {
  int t = blockIdx.x * blockDim.x + threadIdx.x;
  if (t < 6 * 36 * 32) {                       // W_qkv: 216 tiles
    int tile = t >> 5, lane = t & 31;
    int kt = tile / 36, nt = tile % 36;
    int n  = nt * 16 + (lane & 15);
    int kb = kt * 32 + ((lane >> 4) << 4);
    _Float16* dst = wq + ((size_t)t << 4);
    #pragma unroll
    for (int j = 0; j < 16; ++j) dst[j] = (_Float16)Wqkv[(size_t)(kb + j) * C3_ + n];
  } else if (t < 6 * 36 * 32 + 6 * 12 * 32) {  // W_proj: 72 tiles
    int t2 = t - 6 * 36 * 32;
    int tile = t2 >> 5, lane = t2 & 31;
    int kt = tile / 12, nt = tile % 12;
    int n  = nt * 16 + (lane & 15);
    int kb = kt * 32 + ((lane >> 4) << 4);
    _Float16* dst = wp + ((size_t)t2 << 4);
    #pragma unroll
    for (int j = 0; j < 16; ++j) dst[j] = (_Float16)Wproj[(size_t)(kb + j) * C_ + n];
  }
}

// ---- kernel 3: combo[wm][h][p] = bias[h][p] + mask[wm][p] ------------------
__global__ void combo_kernel(const float* __restrict__ biasg, const float* __restrict__ maskg,
                             float* __restrict__ combo) {
  int i = blockIdx.x * blockDim.x + threadIdx.x;
  if (i >= NMASK_ * H_ * N_TOK * N_TOK) return;
  int p  = i & 4095;
  int h  = (i >> 12) % H_;
  int wm = i / (H_ * 4096);
  combo[i] = biasg[h * 4096 + p] + maskg[(size_t)wm * 4096 + p];
}

// ---- kernel 4: fused window attention, one block per window -----------------
__global__ __launch_bounds__(256)
void swin_attn_kernel(const float* __restrict__ x,     const float* __restrict__ maskg,
                      const float* __restrict__ Wqkv,  const float* __restrict__ bqkv,
                      const float* __restrict__ Wproj, const float* __restrict__ bproj,
                      const float* __restrict__ tau,   const float* __restrict__ biasg,
                      const _Float16* __restrict__ wq, const _Float16* __restrict__ wp,
                      const float* __restrict__ combo, int usePack, int useCombo,
                      float* __restrict__ outg) {
  __shared__ __align__(16) _Float16 smA [N_TOK * LDA_];   // tokens f16 -> attn output f16
  __shared__ __align__(16) _Float16 smQ [N_TOK * LDQ_];   // q,k f16  -> final f32 (aliased)
  __shared__ __align__(16) _Float16 smVT[C_ * LDVT_];     // v transposed [hd][token]
  __shared__ __align__(16) _Float16 smP [8 * 16 * 64];    // per-wave softmax tiles

  const int b    = blockIdx.x;
  const int tid  = threadIdx.x;
  // wave id as an SGPR so wave-uniform control flow / addressing stays scalar
  const int w    = __builtin_amdgcn_readfirstlane(tid >> 5);
  const int lane = tid & 31;

  // Phase 1: tokens [n][c] = x[b][c][n], f32 -> f16 (float4 global reads)
  const float* xb = x + (size_t)b * (C_ * N_TOK);
  for (int i4 = tid; i4 < (C_ * N_TOK) / 4; i4 += 256) {
    float4 v = ((const float4*)xb)[i4];
    int i = i4 << 2;
    int c = i >> 6, n = i & 63;
    smA[(n + 0) * LDA_ + c] = (_Float16)v.x;
    smA[(n + 1) * LDA_ + c] = (_Float16)v.y;
    smA[(n + 2) * LDA_ + c] = (_Float16)v.z;
    smA[(n + 3) * LDA_ + c] = (_Float16)v.w;
  }
  __syncthreads();

  // Phase 2: QKV GEMM [64x192]@[192x576], mt-major per wave: A fragments cached in regs.
  // wave w: mt = w&3, nt in [ntBase, ntBase+18); first nQK tiles are q/k, rest are v.
  {
    const int mt = w & 3;
    const int ntBase = (w >> 2) * 18;          // 0 or 18 (SGPR)
    const int nQK = (ntBase == 0) ? 18 : 6;    // tiles with nt < 24 (SGPR)
    v16h aA[6];
    #pragma unroll
    for (int kt = 0; kt < 6; ++kt) aA[kt] = ld_a16(smA, LDA_, mt * 16, kt * 32, lane);

    const int mbase = mt * 16 + ((lane >> 4) << 3);
    if (usePack) {
      for (int i = 0; i < 18; ++i) {
        int nt = ntBase + i;
        v8f acc = {};
        #pragma unroll
        for (int kt = 0; kt < 6; ++kt) {
          v16h bb = *(const v16h*)(wq + ((size_t)((kt * 36 + nt) * 32 + lane) << 4));
          acc = __builtin_amdgcn_wmma_f32_16x16x32_f16(false, aA[kt], false, bb, (short)0, acc, false, false);
        }
        float bv = bqkv[nt * 16 + (lane & 15)];
        int col  = nt * 16 + (lane & 15);
        if (i < nQK) {                      // scalar condition: q/k tiles
          #pragma unroll
          for (int r = 0; r < 8; ++r)
            smQ[(mbase + r) * LDQ_ + col] = (_Float16)(acc[r] + bv);
        } else {                            // v tiles: transposed, contiguous packed store
          v8h pk;
          #pragma unroll
          for (int r = 0; r < 8; ++r) pk[r] = (_Float16)(acc[r] + bv);
          *(v8h*)(smVT + (col - 2 * C_) * LDVT_ + mbase) = pk;
        }
      }
    } else {
      for (int i = 0; i < 18; ++i) {
        int nt = ntBase + i;
        v8f acc = {};
        #pragma unroll
        for (int kt = 0; kt < 6; ++kt) {
          v16h bb = ld_b32g(Wqkv, C3_, kt * 32, nt * 16, lane);
          acc = __builtin_amdgcn_wmma_f32_16x16x32_f16(false, aA[kt], false, bb, (short)0, acc, false, false);
        }
        float bv = bqkv[nt * 16 + (lane & 15)];
        int col  = nt * 16 + (lane & 15);
        if (i < nQK) {
          #pragma unroll
          for (int r = 0; r < 8; ++r)
            smQ[(mbase + r) * LDQ_ + col] = (_Float16)(acc[r] + bv);
        } else {
          v8h pk;
          #pragma unroll
          for (int r = 0; r < 8; ++r) pk[r] = (_Float16)(acc[r] + bv);
          *(v8h*)(smVT + (col - 2 * C_) * LDVT_ + mbase) = pk;
        }
      }
    }
  }
  __syncthreads();

  // Phase 3: l2-normalize q and k rows per head; fold 1/clip(tau) into q
  for (int i = 0; i < 3; ++i) {
    int task = tid + (i << 8);           // 0..767 = {q,k} x 6 heads x 64 rows
    int qk   = task / 384;
    int rem  = task - qk * 384;
    int h = rem >> 6, n = rem & 63;
    _Float16* row = smQ + n * LDQ_ + qk * C_ + h * HD_;
    v8h d[4];
    #pragma unroll
    for (int j = 0; j < 4; ++j) d[j] = *(const v8h*)(row + 8 * j);
    float s = 0.f;
    #pragma unroll
    for (int j = 0; j < 4; ++j)
      #pragma unroll
      for (int e = 0; e < 8; ++e) { float v = (float)d[j][e]; s += v * v; }
    float scale = 1.0f / fmaxf(sqrtf(s), 1e-12f);
    if (qk == 0) scale *= 1.0f / fmaxf(tau[h], 0.01f);
    #pragma unroll
    for (int j = 0; j < 4; ++j) {
      #pragma unroll
      for (int e = 0; e < 8; ++e) d[j][e] = (_Float16)((float)d[j][e] * scale);
      *(v8h*)(row + 8 * j) = d[j];
    }
  }
  __syncthreads();

  // Phase 4: attention. 24 jobs = (head, m-tile) over 8 waves; h, mt are SGPRs.
  const float* maskb  = maskg + (size_t)(b % NMASK_) * (N_TOK * N_TOK);
  const float* combob = combo + ((size_t)((b % NMASK_) * H_) << 12);
  _Float16* smPw = smP + w * (16 * 64);
  for (int i = 0; i < 3; ++i) {
    int job = w + (i << 3);
    int h = job >> 2, mt = job & 3;

    v16h aq = ld_a16(smQ, LDQ_, mt * 16, h * HD_, lane);
    v8f s[4];
    #pragma unroll
    for (int nt = 0; nt < 4; ++nt) {
      v16h bk = ld_bt16(smQ, LDQ_, C_ + h * HD_, nt * 16, lane);
      v8f z = {};
      s[nt] = __builtin_amdgcn_wmma_f32_16x16x32_f16(false, aq, false, bk, (short)0, z, false, false);
    }
    int mbase = mt * 16 + ((lane >> 4) << 3);
    if (useCombo) {
      const float* cb = combob + ((size_t)h << 12);
      #pragma unroll
      for (int r = 0; r < 8; ++r) {
        int m = mbase + r;
        #pragma unroll
        for (int nt = 0; nt < 4; ++nt)
          s[nt][r] += cb[m * 64 + nt * 16 + (lane & 15)];
      }
    } else {
      const float* biash = biasg + (size_t)h * (N_TOK * N_TOK);
      #pragma unroll
      for (int r = 0; r < 8; ++r) {
        int m = mbase + r;
        #pragma unroll
        for (int nt = 0; nt < 4; ++nt) {
          int n = nt * 16 + (lane & 15);
          s[nt][r] += biash[m * 64 + n] + maskb[m * 64 + n];
        }
      }
    }
    // softmax: each row lives in one 16-lane half x 4 accumulators
    #pragma unroll
    for (int r = 0; r < 8; ++r) {
      float mx = fmaxf(fmaxf(s[0][r], s[1][r]), fmaxf(s[2][r], s[3][r]));
      #pragma unroll
      for (int d = 1; d < 16; d <<= 1) mx = fmaxf(mx, __shfl_xor(mx, d, 32));
      float sum = 0.f;
      #pragma unroll
      for (int nt = 0; nt < 4; ++nt) { float e = __expf(s[nt][r] - mx); s[nt][r] = e; sum += e; }
      #pragma unroll
      for (int d = 1; d < 16; d <<= 1) sum += __shfl_xor(sum, d, 32);
      float inv = 1.0f / sum;
      #pragma unroll
      for (int nt = 0; nt < 4; ++nt) s[nt][r] *= inv;
    }
    // stage P as f16 in per-wave LDS to re-fragment C-layout -> A-layout
    #pragma unroll
    for (int r = 0; r < 8; ++r) {
      int m = ((lane >> 4) << 3) + r;
      #pragma unroll
      for (int nt = 0; nt < 4; ++nt)
        smPw[m * 64 + nt * 16 + (lane & 15)] = (_Float16)s[nt][r];
    }
    // P[16x64] @ V[64x32]  (V from transposed buffer; A fragments shared across nh)
    v16h ap0 = ld_a16(smPw, 64, 0, 0,  lane);
    v16h ap1 = ld_a16(smPw, 64, 0, 32, lane);
    #pragma unroll
    for (int nh = 0; nh < 2; ++nh) {
      v8f o = {};
      v16h bv0 = ld_bt16(smVT, LDVT_, 0,  h * HD_ + nh * 16, lane);
      v16h bv1 = ld_bt16(smVT, LDVT_, 32, h * HD_ + nh * 16, lane);
      o = __builtin_amdgcn_wmma_f32_16x16x32_f16(false, ap0, false, bv0, (short)0, o, false, false);
      o = __builtin_amdgcn_wmma_f32_16x16x32_f16(false, ap1, false, bv1, (short)0, o, false, false);
      int col = h * HD_ + nh * 16 + (lane & 15);
      #pragma unroll
      for (int r = 0; r < 8; ++r)
        smA[(mt * 16 + ((lane >> 4) << 3) + r) * LDA_ + col] = (_Float16)o[r];
    }
  }
  __syncthreads();

  // Phase 5: projection GEMM [64x192]@[192x192], mt-major per wave, A cached in regs.
  float* smF = (float*)smQ;   // q,k dead; reuse as f32 staging
  {
    const int mt = w & 3;
    const int ntBase = (w >> 2) * 6;
    v16h aA[6];
    #pragma unroll
    for (int kt = 0; kt < 6; ++kt) aA[kt] = ld_a16(smA, LDA_, mt * 16, kt * 32, lane);

    const int mbase = mt * 16 + ((lane >> 4) << 3);
    if (usePack) {
      for (int i = 0; i < 6; ++i) {
        int nt = ntBase + i;
        v8f acc = {};
        #pragma unroll
        for (int kt = 0; kt < 6; ++kt) {
          v16h bb = *(const v16h*)(wp + ((size_t)((kt * 12 + nt) * 32 + lane) << 4));
          acc = __builtin_amdgcn_wmma_f32_16x16x32_f16(false, aA[kt], false, bb, (short)0, acc, false, false);
        }
        float bv = bproj[nt * 16 + (lane & 15)];
        int col  = nt * 16 + (lane & 15);
        #pragma unroll
        for (int r = 0; r < 8; ++r)
          smF[(mbase + r) * LDF_ + col] = acc[r] + bv;
      }
    } else {
      for (int i = 0; i < 6; ++i) {
        int nt = ntBase + i;
        v8f acc = {};
        #pragma unroll
        for (int kt = 0; kt < 6; ++kt) {
          v16h bb = ld_b32g(Wproj, C_, kt * 32, nt * 16, lane);
          acc = __builtin_amdgcn_wmma_f32_16x16x32_f16(false, aA[kt], false, bb, (short)0, acc, false, false);
        }
        float bv = bproj[nt * 16 + (lane & 15)];
        int col  = nt * 16 + (lane & 15);
        #pragma unroll
        for (int r = 0; r < 8; ++r)
          smF[(mbase + r) * LDF_ + col] = acc[r] + bv;
      }
    }
  }
  __syncthreads();

  // Phase 6: coalesced transposed store  out[b][c][n] = res[n][c]  (float4 writes)
  float* outb = outg + (size_t)b * (C_ * N_TOK);
  for (int i4 = tid; i4 < (C_ * N_TOK) / 4; i4 += 256) {
    int i = i4 << 2;
    int c = i >> 6, n = i & 63;
    float4 v;
    v.x = smF[(n + 0) * LDF_ + c];
    v.y = smF[(n + 1) * LDF_ + c];
    v.z = smF[(n + 2) * LDF_ + c];
    v.w = smF[(n + 3) * LDF_ + c];
    ((float4*)outb)[i4] = v;
  }
}

extern "C" void kernel_launch(void* const* d_in, const int* in_sizes, int n_in,
                              void* d_out, int out_size, void* d_ws, size_t ws_size,
                              hipStream_t stream) {
  const float* x     = (const float*)d_in[0];
  const float* maskg = (const float*)d_in[1];
  const float* Wqkv  = (const float*)d_in[2];
  const float* bqkv  = (const float*)d_in[3];
  const float* Wproj = (const float*)d_in[4];
  const float* bproj = (const float*)d_in[5];
  const float* mW1   = (const float*)d_in[6];
  const float* mb1   = (const float*)d_in[7];
  const float* mW2   = (const float*)d_in[8];
  const float* mb2   = (const float*)d_in[9];
  const float* tau   = (const float*)d_in[10];
  float* outg = (float*)d_out;

  char* ws = (char*)d_ws;
  float*    biasg = (float*)   (ws + WS_BIAS_OFF);
  _Float16* wq    = (_Float16*)(ws + WS_WQ_OFF);
  _Float16* wp    = (_Float16*)(ws + WS_WP_OFF);
  float*    combo = (float*)   (ws + WS_COMBO_OFF);
  int usePack  = (ws_size >= (size_t)WS_COMBO_OFF) ? 1 : 0;
  int useCombo = (ws_size >= (size_t)WS_TOTAL)     ? 1 : 0;

  crpb_bias_kernel<<<(N_TOK * N_TOK + 255) / 256, 256, 0, stream>>>(mW1, mb1, mW2, mb2, biasg);
  if (usePack)
    pack_weights_kernel<<<(6 * 36 * 32 + 6 * 12 * 32 + 255) / 256, 256, 0, stream>>>(Wqkv, Wproj, wq, wp);
  if (useCombo)
    combo_kernel<<<(NMASK_ * H_ * N_TOK * N_TOK + 255) / 256, 256, 0, stream>>>(biasg, maskg, combo);

  swin_attn_kernel<<<BW_, 256, 0, stream>>>(x, maskg, Wqkv, bqkv, Wproj, bproj, tau,
                                            biasg, wq, wp, combo, usePack, useCombo, outg);
}
